// complexNDM_37014028157190
// MI455X (gfx1250) — matrix operand order repaired and checked by reading, not compiled, not run on hardware
//
#include <hip/hip_runtime.h>
#include <hip/hip_bf16.h>

// ---------------- problem constants ----------------
#define BB 16
#define TT 4096
#define HH 256
#define OO 64
#define MT (BB * TT)   // 65536 rows for the fu path
#define SEG 64         // scan segments
#define CHK 64         // timesteps per segment (SEG*CHK == TT)

typedef _Float16 h8   __attribute__((ext_vector_type(8)));
typedef _Float16 v16h __attribute__((ext_vector_type(16)));
typedef float    v8f  __attribute__((ext_vector_type(8)));

// LDS tile pitch in halves: 40 halves = 80 bytes = 5*16B (keeps b128 reads aligned)
#define LPITCH 40

// ---- async global->LDS availability (compile-safe probes) ----
#if defined(__has_builtin)
#if __has_builtin(__builtin_amdgcn_global_load_async_to_lds_b128)
#define HAVE_ASYNC_LDS 1
#endif
#endif
#ifndef HAVE_ASYNC_LDS
#define HAVE_ASYNC_LDS 0
#endif

#if HAVE_ASYNC_LDS
// clang declares the builtin as (v4i addrspace(1)*, v4i addrspace(3)*, imm, imm)
typedef int v4i_vs __attribute__((vector_size(4 * sizeof(int))));
typedef __attribute__((address_space(1))) v4i_vs* g_v4i_ptr;
typedef __attribute__((address_space(3))) v4i_vs* l_v4i_ptr;

__device__ __forceinline__ void async_ld_b128(const _Float16* g, _Float16* l) {
    __builtin_amdgcn_global_load_async_to_lds_b128((g_v4i_ptr)g, (l_v4i_ptr)l, 0, 0);
}
__device__ __forceinline__ void wait_async0() {
#if __has_builtin(__builtin_amdgcn_s_wait_asynccnt)
    __builtin_amdgcn_s_wait_asynccnt(0);
#else
    asm volatile("s_wait_asynccnt 0x0" ::: "memory");
#endif
}
#endif

// ---- fragment gathers matching the CDNA5 16-bit WMMA VGPR layouts ----
// A (16x32, MxK): lane l holds row m=l&15; lanes 0-15: K=0..7 then 16..23,
// lanes 16-31: K=8..15 then 24..31  (two aligned 16B reads per lane)
__device__ inline v16h load_a_frag(const _Float16* lsA, int mbase, int lane) {
    int m  = mbase + (lane & 15);
    int hi = lane >> 4;
    const _Float16* p = lsA + m * LPITCH;
    h8 lo = *(const h8*)(p + 8 * hi);
    h8 h2 = *(const h8*)(p + 16 + 8 * hi);
    return __builtin_shufflevector(lo, h2, 0,1,2,3,4,5,6,7,8,9,10,11,12,13,14,15);
}
// B (32x16, KxN), staged transposed in LDS as [n][k]: lane l holds col n=l&15;
// lanes 0-15: K=0..7 then 16..23, lanes 16-31: K=8..15 then 24..31
__device__ inline v16h load_b_frag(const _Float16* lsB, int nbase, int lane) {
    int n  = nbase + (lane & 15);
    int hi = lane >> 4;
    const _Float16* p = lsB + n * LPITCH + 8 * hi;
    h8 lo = *(const h8*)(p);
    h8 h2 = *(const h8*)(p + 16);
    return __builtin_shufflevector(lo, h2, 0,1,2,3,4,5,6,7,8,9,10,11,12,13,14,15);
}

__device__ inline float swishf(float x) { return x / (1.0f + __expf(-x)); }

// AMODE: 0 = A is f16 row-major, 1 = A is f32 row-major
// EMODE: 0 = swish -> f16 buffer (ld = N)
//        1 = complex h0: out f32 interleaved at hid[2*(m*HH+n)+comp]
//        2 = complex ut: m = b*TT+t -> hid[2*(((t+1)*BB+b)*HH+n)+comp]
//        3 = readout:    m = t*BB+b -> out[(b*TT+t)*OO+n]
// Wt is f16 pre-transposed: [N][K], row stride K.
template <int AMODE, int EMODE>
__global__ __launch_bounds__(256)
void gemm_wmma(const void* __restrict__ Av, int lda,
               const _Float16* __restrict__ Wt,
               const float* __restrict__ bias,
               void* __restrict__ Outv,
               int M, int N, int K, int comp) {
    __shared__ __align__(16) _Float16 lsA[64 * LPITCH];
    __shared__ __align__(16) _Float16 lsB[128 * LPITCH];

    const int tid  = threadIdx.x;
    const int lane = tid & 31;
    const int wid  = tid >> 5;
    const int waveM = (wid & 1) * 32;   // 2 waves along M -> 64 rows
    const int waveN = (wid >> 1) * 32;  // 4 waves along N -> 128 cols
    const int m0 = blockIdx.y * 64;
    const int n0 = blockIdx.x * 128;

    v8f acc[2][2] = {};

    for (int kt = 0; kt < K; kt += 32) {
        // ---- stage A tile (64x32): 256 threads x 8 halves ----
        {
            int row = tid >> 2;
            int col = (tid & 3) * 8;
            _Float16* dst = lsA + row * LPITCH + col;
            int gm = m0 + row;
            if (AMODE == 0) {
#if HAVE_ASYNC_LDS
                if (gm < M) {
                    const _Float16* a = (const _Float16*)Av + (size_t)gm * lda + kt + col;
                    async_ld_b128(a, dst);
                }
#else
                if (gm < M) {
                    const _Float16* a = (const _Float16*)Av + (size_t)gm * lda + kt + col;
                    *(h8*)dst = *(const h8*)a;
                } else {
                    for (int q = 0; q < 8; ++q) dst[q] = (_Float16)0.0f;
                }
#endif
            } else {
                if (gm < M) {
                    const float* a = (const float*)Av + (size_t)gm * lda + kt + col;
                    float4 f0 = *(const float4*)(a);
                    float4 f1 = *(const float4*)(a + 4);
                    dst[0] = (_Float16)f0.x; dst[1] = (_Float16)f0.y;
                    dst[2] = (_Float16)f0.z; dst[3] = (_Float16)f0.w;
                    dst[4] = (_Float16)f1.x; dst[5] = (_Float16)f1.y;
                    dst[6] = (_Float16)f1.z; dst[7] = (_Float16)f1.w;
                } else {
                    for (int q = 0; q < 8; ++q) dst[q] = (_Float16)0.0f;
                }
            }
        }
        // ---- stage B tile (128 x 32 halves), pre-transposed f16 weights ----
        for (int rep = 0; rep < 2; ++rep) {
            int e  = rep * 256 + tid;
            int nn = e >> 2;           // 0..127
            int kk = (e & 3) * 8;      // 0,8,16,24
            _Float16* dst = lsB + nn * LPITCH + kk;
#if HAVE_ASYNC_LDS
            if (n0 + nn < N) {
                const _Float16* w = Wt + (size_t)(n0 + nn) * K + kt + kk;
                async_ld_b128(w, dst);
            }
#else
            if (n0 + nn < N) {
                const _Float16* w = Wt + (size_t)(n0 + nn) * K + kt + kk;
                *(h8*)dst = *(const h8*)w;
            } else {
                for (int q = 0; q < 8; ++q) dst[q] = (_Float16)0.0f;
            }
#endif
        }
#if HAVE_ASYNC_LDS
        wait_async0();
#endif
        __syncthreads();

        v16h aF0 = load_a_frag(lsA, waveM,      lane);
        v16h aF1 = load_a_frag(lsA, waveM + 16, lane);
        v16h bF0 = load_b_frag(lsB, waveN,      lane);
        v16h bF1 = load_b_frag(lsB, waveN + 16, lane);

        acc[0][0] = __builtin_amdgcn_wmma_f32_16x16x32_f16(false, aF0, false, bF0,
                                                           (short)0, acc[0][0], false, false);
        acc[0][1] = __builtin_amdgcn_wmma_f32_16x16x32_f16(false, aF0, false, bF1,
                                                           (short)0, acc[0][1], false, false);
        acc[1][0] = __builtin_amdgcn_wmma_f32_16x16x32_f16(false, aF1, false, bF0,
                                                           (short)0, acc[1][0], false, false);
        acc[1][1] = __builtin_amdgcn_wmma_f32_16x16x32_f16(false, aF1, false, bF1,
                                                           (short)0, acc[1][1], false, false);
        __syncthreads();
    }

    // ---- epilogue: C/D layout: VGPR v, lanes 0-15 -> M=v, lanes 16-31 -> M=8+v; N=lane&15
    for (int i = 0; i < 2; ++i) {
        for (int j = 0; j < 2; ++j) {
            int tileM = m0 + waveM + i * 16 + (lane >> 4) * 8;
            int n     = n0 + waveN + j * 16 + (lane & 15);
            if (n >= N) continue;
            float bval = bias ? bias[n] : 0.0f;
            for (int v = 0; v < 8; ++v) {
                int m = tileM + v;
                if (m >= M) continue;
                float val = acc[i][j][v] + bval;
                if (EMODE == 0) {
                    ((_Float16*)Outv)[(size_t)m * N + n] = (_Float16)swishf(val);
                } else if (EMODE == 1) {
                    ((float*)Outv)[2 * ((size_t)m * HH + n) + comp] = val;
                } else if (EMODE == 2) {
                    int t = m & (TT - 1);
                    int b = m >> 12;
                    ((float*)Outv)[2 * ((size_t)((t + 1) * BB + b) * HH + n) + comp] = val;
                } else { // EMODE == 3
                    int t = m >> 4;
                    int b = m & (BB - 1);
                    ((float*)Outv)[((size_t)b * TT + t) * OO + n] = val;
                }
            }
        }
    }
}

// ---- weight prep: f32 [K,N] -> f16 transposed [N,K] ----
__global__ void transpose_cvt(const float* __restrict__ src, _Float16* __restrict__ dst,
                              int K, int N) {
    int e = blockIdx.x * 256 + threadIdx.x;
    if (e < K * N) {
        int k = e / N, n = e % N;
        dst[(size_t)n * K + k] = (_Float16)src[e];
    }
}
// Sign-folded, interleaved, transposed readout weights: wt[o][2h]=kr, wt[o][2h+1]=-ki
__global__ void prep_wct(const float* __restrict__ kr, const float* __restrict__ ki,
                         _Float16* __restrict__ wt) {
    int e = blockIdx.x * 256 + threadIdx.x;
    if (e < HH * OO) {
        int h = e / OO, o = e % OO;
        wt[(size_t)o * (2 * HH) + 2 * h]     = (_Float16)kr[e];
        wt[(size_t)o * (2 * HH) + 2 * h + 1] = (_Float16)(-ki[e]);
    }
}

__device__ inline void eigval(const float* v_log, const float* theta_log, int h,
                              float& ar, float& ai) {
    int hw  = (h < HH / 2) ? h : h - HH / 2;
    float sgn = (h < HH / 2) ? 1.0f : -1.0f;   // conjugate pair in second half
    float mag = __expf(-__expf(v_log[hw]));
    float th  = __expf(theta_log[hw]);
    ar = mag * __cosf(th);
    ai = sgn * mag * __sinf(th);
}

// ---- segmented scan, pass A: zero-seeded local scans (parallel over S*B*H) ----
__global__ __launch_bounds__(256)
void scanA(const float* __restrict__ v_log, const float* __restrict__ theta_log,
           float* __restrict__ hid) {
    int h = threadIdx.x;
    int s = blockIdx.x >> 4;
    int b = blockIdx.x & (BB - 1);
    float ar, ai;
    eigval(v_log, theta_log, h, ar, ai);
    float hr = 0.0f, hi = 0.0f;
    for (int j = 0; j < CHK; ++j) {
        int t = s * CHK + j;
        size_t idx = 2 * ((size_t)((t + 1) * BB + b) * HH + h);
        float ur = hid[idx], ui = hid[idx + 1];
        float nr = ar * hr - ai * hi + ur;
        float ni = ar * hi + ai * hr + ui;
        hr = nr; hi = ni;
        hid[idx] = nr; hid[idx + 1] = ni;
    }
}

// ---- pass B: combine segment carries; carry_0 = h0, carry_{s+1} = A^CHK*carry_s + local_last_s
__global__ __launch_bounds__(256)
void scanB(const float* __restrict__ v_log, const float* __restrict__ theta_log,
           const float* __restrict__ hid, float* __restrict__ carr) {
    int id = blockIdx.x * 256 + threadIdx.x;   // 0 .. B*H-1
    int b = id >> 8;
    int h = id & (HH - 1);
    float ar, ai;
    eigval(v_log, theta_log, h, ar, ai);
    // Ac = A^CHK via 6 complex squarings (CHK = 64)
    float cr2 = ar, ci2 = ai;
    for (int q = 0; q < 6; ++q) {
        float tr = cr2 * cr2 - ci2 * ci2;
        ci2 = 2.0f * cr2 * ci2;
        cr2 = tr;
    }
    float cr = hid[2 * ((size_t)b * HH + h)];       // h0
    float ci = hid[2 * ((size_t)b * HH + h) + 1];
    for (int s = 0; s < SEG; ++s) {
        size_t cidx = 2 * (((size_t)s * BB + b) * HH + h);
        carr[cidx] = cr; carr[cidx + 1] = ci;
        int tl = s * CHK + CHK - 1;
        size_t idx = 2 * ((size_t)((tl + 1) * BB + b) * HH + h);
        float lr = hid[idx], li = hid[idx + 1];
        float nr = cr2 * cr - ci2 * ci + lr;
        float ni = cr2 * ci + ci2 * cr + li;
        cr = nr; ci = ni;
    }
}

// ---- pass C: h_t = local + A^(j+1)*carry; also emit f16 interleaved rows for readout
__global__ __launch_bounds__(256)
void scanC(const float* __restrict__ v_log, const float* __restrict__ theta_log,
           float* __restrict__ hid, const float* __restrict__ carr,
           _Float16* __restrict__ bufR) {
    int h = threadIdx.x;
    int s = blockIdx.x >> 4;
    int b = blockIdx.x & (BB - 1);
    float ar, ai;
    eigval(v_log, theta_log, h, ar, ai);
    size_t cidx = 2 * (((size_t)s * BB + b) * HH + h);
    float cr = carr[cidx], ci = carr[cidx + 1];
    float pr = ar, pi = ai;   // A^(j+1)
    for (int j = 0; j < CHK; ++j) {
        int t = s * CHK + j;
        size_t idx = 2 * ((size_t)((t + 1) * BB + b) * HH + h);
        float nr = hid[idx]     + pr * cr - pi * ci;
        float ni = hid[idx + 1] + pr * ci + pi * cr;
        hid[idx] = nr; hid[idx + 1] = ni;
        size_t ridx = (size_t)(t * BB + b) * (2 * HH) + 2 * h;
        bufR[ridx]     = (_Float16)nr;
        bufR[ridx + 1] = (_Float16)ni;
        float tr = pr * ar - pi * ai;
        pi = pr * ai + pi * ar;
        pr = tr;
    }
}

static inline dim3 gemm_grid(int M, int N) {
    return dim3((unsigned)((N + 127) / 128), (unsigned)((M + 63) / 64), 1);
}

extern "C" void kernel_launch(void* const* d_in, const int* in_sizes, int n_in,
                              void* d_out, int out_size, void* d_ws, size_t ws_size,
                              hipStream_t stream) {
    const float* x0        = (const float*)d_in[0];
    const float* u         = (const float*)d_in[1];
    const float* v_log     = (const float*)d_in[2];
    const float* theta_log = (const float*)d_in[3];
    const float* f0_W      = (const float*)d_in[4];
    const float* f0_b      = (const float*)d_in[5];
    const float* f0_kr     = (const float*)d_in[6];
    const float* f0_ki     = (const float*)d_in[7];
    const float* f0_br     = (const float*)d_in[8];
    const float* f0_bi     = (const float*)d_in[9];
    const float* fu_W      = (const float*)d_in[10];
    const float* fu_b      = (const float*)d_in[11];
    const float* fu_kr     = (const float*)d_in[12];
    const float* fu_ki     = (const float*)d_in[13];
    const float* fu_br     = (const float*)d_in[14];
    const float* fu_bi     = (const float*)d_in[15];
    const float* C_kr      = (const float*)d_in[16];
    const float* C_ki      = (const float*)d_in[17];

    float* outp = (float*)d_out;                       // [B,T,O] f32
    float* hid  = outp + (size_t)BB * TT * OO;         // [T+1,B,H] complex interleaved

    // workspace layout (halves):
    //  bufR  : [MT, 2H] f16   (64 MB) -- first half doubles as buf0, second as buf1
    //  wt    : 8 transposed f16 [256,256] weights + readout [64,512]
    //  carr  : [SEG,B,H] complex f32 (2 MB)
    _Float16* bufR = (_Float16*)d_ws;
    _Float16* buf0 = bufR;
    _Float16* buf1 = bufR + (size_t)MT * HH;
    _Float16* wt   = bufR + (size_t)MT * (2 * HH);
    _Float16* wt_f0W0 = wt;
    _Float16* wt_f0W1 = wt + (size_t)1 * HH * HH;
    _Float16* wt_fuW0 = wt + (size_t)2 * HH * HH;
    _Float16* wt_fuW1 = wt + (size_t)3 * HH * HH;
    _Float16* wt_f0kr = wt + (size_t)4 * HH * HH;
    _Float16* wt_f0ki = wt + (size_t)5 * HH * HH;
    _Float16* wt_fukr = wt + (size_t)6 * HH * HH;
    _Float16* wt_fuki = wt + (size_t)7 * HH * HH;
    _Float16* wt_C    = wt + (size_t)8 * HH * HH;          // [OO, 2H]
    float*    carr    = (float*)(wt_C + (size_t)OO * 2 * HH);

    // ---- weight prep (tiny, L2-resident afterwards) ----
    const int wg = (HH * HH + 255) / 256;
    transpose_cvt<<<wg, 256, 0, stream>>>(f0_W,                     wt_f0W0, HH, HH);
    transpose_cvt<<<wg, 256, 0, stream>>>(f0_W + (size_t)HH * HH,   wt_f0W1, HH, HH);
    transpose_cvt<<<wg, 256, 0, stream>>>(fu_W,                     wt_fuW0, HH, HH);
    transpose_cvt<<<wg, 256, 0, stream>>>(fu_W + (size_t)HH * HH,   wt_fuW1, HH, HH);
    transpose_cvt<<<wg, 256, 0, stream>>>(f0_kr,                    wt_f0kr, HH, HH);
    transpose_cvt<<<wg, 256, 0, stream>>>(f0_ki,                    wt_f0ki, HH, HH);
    transpose_cvt<<<wg, 256, 0, stream>>>(fu_kr,                    wt_fukr, HH, HH);
    transpose_cvt<<<wg, 256, 0, stream>>>(fu_ki,                    wt_fuki, HH, HH);
    prep_wct<<<(HH * OO + 255) / 256, 256, 0, stream>>>(C_kr, C_ki, wt_C);

    // ---- f0 path (M = B = 16) ----
    gemm_wmma<1, 0><<<gemm_grid(BB, HH), 256, 0, stream>>>(
        x0, HH, wt_f0W0, f0_b, buf0, BB, HH, HH, 0);
    gemm_wmma<0, 0><<<gemm_grid(BB, HH), 256, 0, stream>>>(
        buf0, HH, wt_f0W1, f0_b + HH, buf1, BB, HH, HH, 0);
    gemm_wmma<0, 1><<<gemm_grid(BB, HH), 256, 0, stream>>>(
        buf1, HH, wt_f0kr, f0_br, hid, BB, HH, HH, 0);
    gemm_wmma<0, 1><<<gemm_grid(BB, HH), 256, 0, stream>>>(
        buf1, HH, wt_f0ki, f0_bi, hid, BB, HH, HH, 1);

    // ---- fu path (M = B*T = 65536) ----
    gemm_wmma<1, 0><<<gemm_grid(MT, HH), 256, 0, stream>>>(
        u, HH, wt_fuW0, fu_b, buf0, MT, HH, HH, 0);
    gemm_wmma<0, 0><<<gemm_grid(MT, HH), 256, 0, stream>>>(
        buf0, HH, wt_fuW1, fu_b + HH, buf1, MT, HH, HH, 0);
    gemm_wmma<0, 2><<<gemm_grid(MT, HH), 256, 0, stream>>>(
        buf1, HH, wt_fukr, fu_br, hid, MT, HH, HH, 0);
    gemm_wmma<0, 2><<<gemm_grid(MT, HH), 256, 0, stream>>>(
        buf1, HH, wt_fuki, fu_bi, hid, MT, HH, HH, 1);

    // ---- segmented diagonal complex scan (latency-hiding, 3 passes) ----
    scanA<<<SEG * BB, HH, 0, stream>>>(v_log, theta_log, hid);
    scanB<<<(BB * HH) / 256, 256, 0, stream>>>(v_log, theta_log, hid, carr);
    scanC<<<SEG * BB, HH, 0, stream>>>(v_log, theta_log, hid, carr, bufR);

    // ---- readout: f16 interleaved hidden rows (K = 2H = 512) ----
    gemm_wmma<0, 3><<<gemm_grid(MT, OO), 256, 0, stream>>>(
        bufR, 2 * HH, wt_C, nullptr, outp, MT, OO, 2 * HH, 0);
}
